// ReciprocalNN_20323785244869
// MI455X (gfx1250) — matrix operand order repaired and checked by reading, not compiled, not run on hardware
//
#include <hip/hip_runtime.h>

// ---- problem constants ----
#define BATCH   16
#define NATOM   512
#define HDIM    128
#define NLAYER  3
#define NMAXG   9
#define NGRID   19           // 2*NMAX+1
#define KTOT    6859         // 19^3
#define KPAD    6864         // 429*16
#define KMAXV   5.0f
#define ETA_C   0.3f
#define EPS3_C  1.0e-5f

// CONV_FACT = 1e10 * e / (4*pi*eps0)
static constexpr double PI_D = 3.14159265358979323846;
static constexpr double CONV_D = 1.0e10 * 1.602176634e-19 / (4.0 * PI_D * 8.8541878128e-12);
#define CONV_F ((float)CONV_D)
#define TWO_PI_F ((float)(2.0 * PI_D))

typedef __attribute__((ext_vector_type(16))) _Float16 v16h;
typedef __attribute__((ext_vector_type(8)))  float    v8f;

// ---------------------------------------------------------------------------
// Convert Wh (L,H,H) f32 row-major [k][n]  ->  f16 transposed [l][n][k]
// ---------------------------------------------------------------------------
__global__ void prep_w_kernel(const float* __restrict__ Wh, _Float16* __restrict__ Wt) {
    int idx = blockIdx.x * 256 + threadIdx.x;
    if (idx >= NLAYER * HDIM * HDIM) return;
    int l = idx / (HDIM * HDIM);
    int rem = idx % (HDIM * HDIM);
    int n = rem / HDIM;
    int k = rem % HDIM;
    Wt[idx] = (_Float16)Wh[l * HDIM * HDIM + k * HDIM + n];
}

// ---------------------------------------------------------------------------
// Per-batch reciprocal vectors: ks = n @ (2*pi*inv(L)), ksnorm = |ks|
// ---------------------------------------------------------------------------
__global__ void k_setup_kernel(const float* __restrict__ lattice,
                               float* __restrict__ ks, float* __restrict__ ksn) {
    int b = blockIdx.y;
    int k = blockIdx.x * blockDim.x + threadIdx.x;
    if (k >= KPAD) return;
    float kx = 0.f, ky = 0.f, kz = 0.f, nrm = 0.f;
    if (k < KTOT) {
        const float* L = lattice + b * 9;
        float a00=L[0],a01=L[1],a02=L[2],a10=L[3],a11=L[4],a12=L[5],a20=L[6],a21=L[7],a22=L[8];
        // cofactors
        float c00 = a11*a22 - a12*a21, c01 = a12*a20 - a10*a22, c02 = a10*a21 - a11*a20;
        float c10 = a02*a21 - a01*a22, c11 = a00*a22 - a02*a20, c12 = a01*a20 - a00*a21;
        float c20 = a01*a12 - a02*a11, c21 = a02*a10 - a00*a12, c22 = a00*a11 - a01*a10;
        float det = a00*c00 + a01*c01 + a02*c02;
        float s = TWO_PI_F / det;
        float nx = (float)(k / (NGRID * NGRID) - NMAXG);
        float ny = (float)((k / NGRID) % NGRID - NMAXG);
        float nz = (float)(k % NGRID - NMAXG);
        // ks_j = 2pi/det * (cof[j][0]*nx + cof[j][1]*ny + cof[j][2]*nz)
        kx = s * (c00 * nx + c01 * ny + c02 * nz);
        ky = s * (c10 * nx + c11 * ny + c12 * nz);
        kz = s * (c20 * nx + c21 * ny + c22 * nz);
        nrm = sqrtf(kx * kx + ky * ky + kz * kz);
    }
    size_t o = (size_t)b * KPAD + k;
    ks[o * 3 + 0] = kx; ks[o * 3 + 1] = ky; ks[o * 3 + 2] = kz;
    ksn[o] = nrm;
}

// ---------------------------------------------------------------------------
// |S(k)|^2 : one wave (32 lanes) per k-vector, atoms staged in LDS
// ---------------------------------------------------------------------------
__global__ __launch_bounds__(256) void sk_kernel(const float* __restrict__ ks,
                                                 const float* __restrict__ pos,
                                                 const float* __restrict__ chg,
                                                 float* __restrict__ sk2) {
    int b = blockIdx.y;
    __shared__ float px[NATOM], py[NATOM], pz[NATOM], qq[NATOM];
    for (int i = threadIdx.x; i < NATOM; i += 256) {
        const float* p = pos + ((size_t)b * NATOM + i) * 3;
        px[i] = p[0]; py[i] = p[1]; pz[i] = p[2];
        qq[i] = chg[(size_t)b * NATOM + i];
    }
    __syncthreads();
    int wave = threadIdx.x >> 5, lane = threadIdx.x & 31;
    int k = blockIdx.x * 8 + wave;
    if (k >= KPAD) return;
    size_t o = (size_t)b * KPAD + k;
    float kx = ks[o * 3 + 0], ky = ks[o * 3 + 1], kz = ks[o * 3 + 2];
    float re = 0.f, im = 0.f;
    for (int a = lane; a < NATOM; a += 32) {
        float d = kx * px[a] + ky * py[a] + kz * pz[a];
        float s, c;
        __sincosf(d, &s, &c);
        re += qq[a] * c;
        im += qq[a] * s;
    }
    for (int m = 16; m >= 1; m >>= 1) {
        re += __shfl_xor(re, m, 32);
        im += __shfl_xor(im, m, 32);
    }
    if (lane == 0) sk2[o] = re * re + im * im;
}

// ---------------------------------------------------------------------------
// MLP: phi(|k|).  8 waves/block, each wave owns 16 sample rows.
// Hidden layers via v_wmma_f32_16x16x32_f16 with f32 accumulate + LN + SiLU.
// ---------------------------------------------------------------------------
__device__ __forceinline__ float silu_f(float v) {
    return v * (1.f / (1.f + __expf(-v)));
}

__global__ __launch_bounds__(256) void mlp_kernel(
    const float* __restrict__ ksn, const _Float16* __restrict__ Wt,
    const float* __restrict__ W_in, const float* __restrict__ b_in,
    const float* __restrict__ g_in, const float* __restrict__ beta_in,
    const float* __restrict__ bh, const float* __restrict__ gh,
    const float* __restrict__ betah,
    const float* __restrict__ W_out, const float* __restrict__ b_out,
    float* __restrict__ phi) {

    __shared__ _Float16 Wlds[HDIM * HDIM];        // 32 KB  (current layer weights, [n][k])
    __shared__ _Float16 Slds[8][16 * HDIM];       // 32 KB  (per-wave 16x128 activations)

    const int wave = threadIdx.x >> 5;
    const int lane = threadIdx.x & 31;
    const int lo = lane & 15;
    const int hi = lane >> 4;
    const size_t rowBase = ((size_t)blockIdx.x * 8 + wave) * 16;

    // ---- layer 0: h = silu(LN(s * W_in + b_in)), elementwise per row ----
    for (int r = 0; r < 16; ++r) {
        float s = ksn[rowBase + r];
        float x[4]; float sum = 0.f, sq = 0.f;
        for (int j = 0; j < 4; ++j) {
            int c = lane + 32 * j;
            x[j] = s * W_in[c] + b_in[c];
            sum += x[j]; sq += x[j] * x[j];
        }
        for (int m = 16; m >= 1; m >>= 1) {
            sum += __shfl_xor(sum, m, 32);
            sq  += __shfl_xor(sq,  m, 32);
        }
        float mean = sum * (1.f / HDIM);
        float var  = sq * (1.f / HDIM) - mean * mean;
        float rstd = rsqrtf(var + 1e-5f);
        for (int j = 0; j < 4; ++j) {
            int c = lane + 32 * j;
            float v = (x[j] - mean) * rstd * g_in[c] + beta_in[c];
            Slds[wave][r * HDIM + c] = (_Float16)silu_f(v);
        }
    }

    // ---- hidden layers: (16x128) @ (128x128) via WMMA ----
    for (int l = 0; l < NLAYER; ++l) {
        __syncthreads();                       // all waves done reading previous Wlds
        const _Float16* wsrc = Wt + (size_t)l * HDIM * HDIM;
        for (int i = threadIdx.x; i < HDIM * HDIM / 2; i += 256)
            ((float*)Wlds)[i] = ((const float*)wsrc)[i];
        __syncthreads();

        // A fragments: ISA 16-bit A layout. lane%16 = M; half h -> K
        v16h A[4];
        for (int kc = 0; kc < 4; ++kc) {
            for (int h2 = 0; h2 < 8; ++h2) {
                int hb = h2 * 2;
                int K = kc * 32 + ((hb < 8) ? 0 : 16) + 8 * hi + (hb & 7);
                A[kc][hb]     = Slds[wave][lo * HDIM + K];
                A[kc][hb + 1] = Slds[wave][lo * HDIM + K + 1];
            }
        }

        v8f acc[8];
        for (int nt = 0; nt < 8; ++nt) {
            v8f c = {};
            for (int kc = 0; kc < 4; ++kc) {
                // B fragments: lane%16 = N; same K layout as A. Wlds is [n][k].
                v16h Bf;
                int n = nt * 16 + lo;
                for (int h2 = 0; h2 < 8; ++h2) {
                    int hb = h2 * 2;
                    int K = kc * 32 + ((hb < 8) ? 0 : 16) + 8 * hi + (hb & 7);
                    Bf[hb]     = Wlds[n * HDIM + K];
                    Bf[hb + 1] = Wlds[n * HDIM + K + 1];
                }
                c = __builtin_amdgcn_wmma_f32_16x16x32_f16(
                        false, A[kc], false, Bf, (short)0, c, false, false);
            }
            acc[nt] = c;
        }

        // bias + LN + SiLU on f32 accumulators; C layout: VGPR r -> row r (+8 if hi),
        // lane%16 -> N%16, tile nt -> N/16.
        for (int r = 0; r < 8; ++r) {
            float vals[8]; float sum = 0.f, sq = 0.f;
            for (int nt = 0; nt < 8; ++nt) {
                int n = nt * 16 + lo;
                float v = acc[nt][r] + bh[l * HDIM + n];
                vals[nt] = v; sum += v; sq += v * v;
            }
            for (int m = 8; m >= 1; m >>= 1) {           // reduce within 16-lane group
                sum += __shfl_xor(sum, m, 16);
                sq  += __shfl_xor(sq,  m, 16);
            }
            float mean = sum * (1.f / HDIM);
            float var  = sq * (1.f / HDIM) - mean * mean;
            float rstd = rsqrtf(var + 1e-5f);
            int mrow = r + 8 * hi;
            for (int nt = 0; nt < 8; ++nt) {
                int n = nt * 16 + lo;
                float v = (vals[nt] - mean) * rstd * gh[l * HDIM + n] + betah[l * HDIM + n];
                Slds[wave][mrow * HDIM + n] = (_Float16)silu_f(v);
            }
        }
    }

    // ---- output layer: phi = h @ W_out + b_out (128-dot per row) ----
    for (int r = 0; r < 16; ++r) {
        float sum = 0.f;
        for (int j = 0; j < 4; ++j) {
            int c = lane + 32 * j;
            sum += (float)Slds[wave][r * HDIM + c] * W_out[c];
        }
        for (int m = 16; m >= 1; m >>= 1) sum += __shfl_xor(sum, m, 32);
        if (lane == 0) phi[rowBase + r] = sum + b_out[0];
    }
}

// ---------------------------------------------------------------------------
// Masked reduction over K + cell term + all outputs
// out layout: [0,B) E_recip | [B,2B) total_energy | [2B, 2B+B*N) per_atom
// ---------------------------------------------------------------------------
__global__ __launch_bounds__(256) void finalize_kernel(
    const float* __restrict__ lattice, const float* __restrict__ ksn,
    const float* __restrict__ phi, const float* __restrict__ sk2,
    const float* __restrict__ total_charge, const float* __restrict__ E_short,
    const float* __restrict__ atomic_E_short, const float* __restrict__ alpha,
    float* __restrict__ out) {

    int b = blockIdx.x;
    __shared__ float red[256];
    __shared__ float sEr, sEc;

    const float* L = lattice + b * 9;
    float a00=L[0],a01=L[1],a02=L[2],a10=L[3],a11=L[4],a12=L[5],a20=L[6],a21=L[7],a22=L[8];
    float det = a00*(a11*a22 - a12*a21) - a01*(a10*a22 - a12*a20) + a02*(a10*a21 - a11*a20);
    float vol = fabsf(det);

    float acc = 0.f;
    for (int k = threadIdx.x; k < KTOT; k += 256) {
        size_t o = (size_t)b * KPAD + k;
        float nrm = ksn[o];
        if (nrm > EPS3_C && nrm < KMAXV) acc += phi[o] * sk2[o];
    }
    red[threadIdx.x] = acc;
    __syncthreads();
    for (int s = 128; s > 0; s >>= 1) {
        if (threadIdx.x < s) red[threadIdx.x] += red[threadIdx.x + s];
        __syncthreads();
    }
    if (threadIdx.x == 0) {
        float Er = CONV_F * red[0] / vol;
        float Q = total_charge[b];
        float Ec = -CONV_F * 0.5f * (float)PI_D / vol / ETA_C * Q * Q * alpha[0];
        out[b] = Er;
        out[BATCH + b] = E_short[b] + Er + Ec;
        sEr = Er; sEc = Ec;
    }
    __syncthreads();
    float pa = (sEr + sEc) * (1.f / NATOM);
    for (int i = threadIdx.x; i < NATOM; i += 256)
        out[2 * BATCH + b * NATOM + i] = atomic_E_short[(size_t)b * NATOM + i] + pa;
}

// ---------------------------------------------------------------------------
extern "C" void kernel_launch(void* const* d_in, const int* in_sizes, int n_in,
                              void* d_out, int out_size, void* d_ws, size_t ws_size,
                              hipStream_t stream) {
    const float* lattice        = (const float*)d_in[0];
    const float* positions      = (const float*)d_in[1];
    const float* charges        = (const float*)d_in[2];
    const float* total_charge   = (const float*)d_in[3];
    const float* E_short        = (const float*)d_in[4];
    const float* atomic_E_short = (const float*)d_in[5];
    const float* W_in           = (const float*)d_in[6];
    const float* b_in           = (const float*)d_in[7];
    const float* g_in           = (const float*)d_in[8];
    const float* beta_in        = (const float*)d_in[9];
    const float* Wh             = (const float*)d_in[10];
    const float* bh             = (const float*)d_in[11];
    const float* gh             = (const float*)d_in[12];
    const float* betah          = (const float*)d_in[13];
    const float* W_out          = (const float*)d_in[14];
    const float* b_out          = (const float*)d_in[15];
    const float* alpha          = (const float*)d_in[16];
    float* out = (float*)d_out;

    // workspace layout (float slots)
    float* ws = (float*)d_ws;
    float*     f_ks   = ws;                                    // B*KPAD*3
    float*     f_norm = f_ks   + (size_t)BATCH * KPAD * 3;     // B*KPAD
    float*     f_sk2  = f_norm + (size_t)BATCH * KPAD;         // B*KPAD
    float*     f_phi  = f_sk2  + (size_t)BATCH * KPAD;         // B*KPAD
    _Float16*  f_wt   = (_Float16*)(f_phi + (size_t)BATCH * KPAD); // 3*H*H halves

    prep_w_kernel<<<(NLAYER * HDIM * HDIM + 255) / 256, 256, 0, stream>>>(Wh, f_wt);

    k_setup_kernel<<<dim3((KPAD + 255) / 256, BATCH), 256, 0, stream>>>(lattice, f_ks, f_norm);

    sk_kernel<<<dim3(KPAD / 8, BATCH), 256, 0, stream>>>(f_ks, positions, charges, f_sk2);

    mlp_kernel<<<(BATCH * KPAD) / 128, 256, 0, stream>>>(
        f_norm, f_wt, W_in, b_in, g_in, beta_in, bh, gh, betah, W_out, b_out, f_phi);

    finalize_kernel<<<BATCH, 256, 0, stream>>>(
        lattice, f_norm, f_phi, f_sk2, total_charge, E_short, atomic_E_short, alpha, out);
}
